// Attention_5841155522607
// MI455X (gfx1250) — compile-verified, matrix operand order checked by reading, not run on hardware
//
#include <hip/hip_runtime.h>
#include <hip/hip_bf16.h>

typedef __attribute__((ext_vector_type(16))) _Float16 v16h;
typedef __attribute__((ext_vector_type(8)))  _Float16 h8;
typedef __attribute__((ext_vector_type(8)))  float    v8f;

#define DIM_   512
#define HEADS_ 8
#define HD_    64
#define B_     4
#define NQ_    4096
#define NK_    1024
#define HW_    64
#define LDW_   520   /* padded LDS row stride (halfs): 1040B -> conflict-free */

// ---------------------------------------------------------------------------
// CDNA5 async copy global->LDS (ASYNCcnt), ISA 08_async_tensor.
// ---------------------------------------------------------------------------
static __device__ __forceinline__ void async_b128(const _Float16* lds,
                                                  const _Float16* g) {
  unsigned loff = (unsigned)(uintptr_t)lds;
  unsigned long long ga = (unsigned long long)(uintptr_t)g;
  asm volatile("global_load_async_to_lds_b128 %0, %1, off"
               :: "v"(loff), "v"(ga) : "memory");
}
#define WAIT_ASYNC(n) asm volatile("s_wait_asynccnt %0" :: "n"(n) : "memory")

// ---------------------------------------------------------------------------
// WMMA fragment helpers (16x16x32 f16 layout, ISA 7.12.2)
// ---------------------------------------------------------------------------
static __device__ __forceinline__ v16h frag16(const _Float16* p) {
  h8 lo = *(const h8*)p;
  h8 hi = *(const h8*)(p + 16);
  v16h r;
#pragma unroll
  for (int i = 0; i < 8; ++i) { r[i] = lo[i]; r[i + 8] = hi[i]; }
  return r;
}
static __device__ __forceinline__ v16h fragRM(const _Float16* base, int ld, int lane) {
  return frag16(base + (size_t)(lane & 15) * ld + ((lane >> 4) << 3));
}
static __device__ __forceinline__ v8f wmma_f16(v16h a, v16h b, v8f c) {
  return __builtin_amdgcn_wmma_f32_16x16x32_f16(false, a, false, b, (short)0, c,
                                                false, false);
}

// ---------------------------------------------------------------------------
// Prep kernels
// ---------------------------------------------------------------------------
__global__ void cvt_f32_to_f16(const float* __restrict__ in,
                               _Float16* __restrict__ out, int n) {
  int i = blockIdx.x * blockDim.x + threadIdx.x;
  if (i < n) out[i] = (_Float16)in[i];
}

__global__ void repack_srw(const float* __restrict__ in,
                           _Float16* __restrict__ out) {
  int t = blockIdx.x * blockDim.x + threadIdx.x;
  if (t >= 4 * DIM_ * DIM_) return;
  int ij  = t >> 18;
  int rem = t & 262143;
  out[t] = (_Float16)in[(size_t)rem * 4 + ij];
}

// ---------------------------------------------------------------------------
// Shared GEMM core: 32(M) x 64(N) wave tile, K unrolled by 64 with ping/pong
// A registers (no rotation movs) and batched B-fragment LDS loads.
// ---------------------------------------------------------------------------
#define GEMM_CORE(A0PTR, A1PTR, KTOT)                                         \
  v8f acc[2][4] = {{{}, {}, {}, {}}, {{}, {}, {}, {}}};                       \
  v16h aP0 = frag16(A0PTR), aP1 = frag16(A1PTR);                              \
  for (int kc = 0; kc < (KTOT); kc += 64) {                                   \
    v16h aQ0 = frag16((A0PTR) + kc + 32);                                     \
    v16h aQ1 = frag16((A1PTR) + kc + 32);                                     \
    v16h bb[4];                                                               \
    _Pragma("unroll")                                                         \
    for (int t = 0; t < 4; ++t)                                               \
      bb[t] = fragRM(Wl + t * 16 * LDW_ + kc, LDW_, lane);                    \
    _Pragma("unroll")                                                         \
    for (int t = 0; t < 4; ++t) {                                             \
      acc[0][t] = wmma_f16(aP0, bb[t], acc[0][t]);                            \
      acc[1][t] = wmma_f16(aP1, bb[t], acc[1][t]);                            \
    }                                                                         \
    if (kc + 64 < (KTOT)) {                                                   \
      aP0 = frag16((A0PTR) + kc + 64);                                        \
      aP1 = frag16((A1PTR) + kc + 64);                                        \
    }                                                                         \
    _Pragma("unroll")                                                         \
    for (int t = 0; t < 4; ++t)                                               \
      bb[t] = fragRM(Wl + t * 16 * LDW_ + kc + 32, LDW_, lane);               \
    _Pragma("unroll")                                                         \
    for (int t = 0; t < 4; ++t) {                                             \
      acc[0][t] = wmma_f16(aQ0, bb[t], acc[0][t]);                            \
      acc[1][t] = wmma_f16(aQ1, bb[t], acc[1][t]);                            \
    }                                                                         \
  }

#define STAGE_W(WPTR, LD)                                                     \
  for (int idx = threadIdx.x; idx < 64 * 64; idx += 256) {                    \
    int r = idx >> 6, c = (idx & 63) << 3;                                    \
    async_b128(Wl + r * LDW_ + c, (WPTR) + (size_t)(nBase + r) * (LD) + c);   \
  }                                                                           \
  WAIT_ASYNC(0);                                                              \
  __syncthreads();

// ---------------------------------------------------------------------------
// WMMA GEMM, f16 out:  O[M][N] = A[M][K] @ W[N][K]^T
// ---------------------------------------------------------------------------
__global__ __launch_bounds__(256) void gemm_f16out(
    const _Float16* __restrict__ A, const _Float16* __restrict__ W,
    _Float16* __restrict__ O, int K, int N) {
  __shared__ __align__(16) _Float16 Wl[64 * LDW_];
  int lane = threadIdx.x & 31, wv = threadIdx.x >> 5;
  int mBase = blockIdx.x * 256 + wv * 32;
  int nBase = blockIdx.y * 64;

  STAGE_W(W, K)

  int koff = (lane >> 4) << 3;
  const _Float16* A0p = A + (size_t)(mBase + (lane & 15)) * K + koff;
  const _Float16* A1p = A0p + (size_t)16 * K;
  GEMM_CORE(A0p, A1p, K)

  int col0 = nBase + (lane & 15);
  int rhi  = (lane >> 4) << 3;
#pragma unroll
  for (int s = 0; s < 2; ++s)
#pragma unroll
    for (int t = 0; t < 4; ++t)
#pragma unroll
      for (int i = 0; i < 8; ++i)
        O[(size_t)(mBase + s * 16 + rhi + i) * N + col0 + t * 16] =
            (_Float16)acc[s][t][i];
}

// ---------------------------------------------------------------------------
// SR conv as 4 gather-GEMMs, f32 out (per-ij weight tile in LDS).
// ---------------------------------------------------------------------------
__global__ __launch_bounds__(256) void conv_gemm(
    const _Float16* __restrict__ F, const _Float16* __restrict__ Wsr,
    float* __restrict__ O) {
  __shared__ __align__(16) _Float16 Wl[64 * LDW_];
  int lane = threadIdx.x & 31, wv = threadIdx.x >> 5;
  int mBase = blockIdx.x * 256 + wv * 32;
  int nBase = blockIdx.y * 64;
  int b  = mBase >> 10;
  int ok0 = (mBase & 1023) + (lane & 15);
  int ok1 = ok0 + 16;
  int koff = (lane >> 4) << 3;
  v8f accT[2][4] = {{{}, {}, {}, {}}, {{}, {}, {}, {}}};

#pragma unroll
  for (int ij = 0; ij < 4; ++ij) {
    __syncthreads();
    const _Float16* wij = Wsr + (size_t)ij * DIM_ * DIM_;
    STAGE_W(wij, DIM_)

    int ii = ij >> 1, jj = ij & 1;
    int pix0 = (2 * (ok0 >> 5) + ii) * HW_ + (2 * (ok0 & 31) + jj);
    int pix1 = (2 * (ok1 >> 5) + ii) * HW_ + (2 * (ok1 & 31) + jj);
    const _Float16* A0p = F + ((size_t)(b * (HW_ * HW_) + pix0)) * DIM_ + koff;
    const _Float16* A1p = F + ((size_t)(b * (HW_ * HW_) + pix1)) * DIM_ + koff;
    {
      GEMM_CORE(A0p, A1p, DIM_)
#pragma unroll
      for (int s = 0; s < 2; ++s)
#pragma unroll
        for (int t = 0; t < 4; ++t)
#pragma unroll
          for (int i = 0; i < 8; ++i)
            accT[s][t][i] += acc[s][t][i];
    }
  }

  int col0 = nBase + (lane & 15);
  int rhi  = (lane >> 4) << 3;
#pragma unroll
  for (int s = 0; s < 2; ++s)
#pragma unroll
    for (int t = 0; t < 4; ++t)
#pragma unroll
      for (int i = 0; i < 8; ++i)
        O[(size_t)(mBase + s * 16 + rhi + i) * DIM_ + col0 + t * 16] =
            accT[s][t][i];
}

// ---------------------------------------------------------------------------
// LayerNorm( xconv + sr_b ) -> f16.  One wave per row.
// ---------------------------------------------------------------------------
__global__ __launch_bounds__(256) void ln_kernel(
    const float* __restrict__ X, const float* __restrict__ srb,
    const float* __restrict__ g, const float* __restrict__ bb,
    _Float16* __restrict__ O) {
  int lane = threadIdx.x & 31;
  int row  = blockIdx.x * 8 + (threadIdx.x >> 5);
  const float* x = X + (size_t)row * DIM_;
  float vals[16], s = 0.f, s2 = 0.f;
#pragma unroll
  for (int i = 0; i < 16; ++i) {
    int c = lane + i * 32;
    float v = x[c] + srb[c];
    vals[i] = v; s += v; s2 += v * v;
  }
#pragma unroll
  for (int off = 1; off < 32; off <<= 1) {
    s  += __shfl_xor(s, off, 32);
    s2 += __shfl_xor(s2, off, 32);
  }
  float mu  = s * (1.f / DIM_);
  float var = s2 * (1.f / DIM_) - mu * mu;
  float r   = rsqrtf(var + 1e-5f);
#pragma unroll
  for (int i = 0; i < 16; ++i) {
    int c = lane + i * 32;
    O[(size_t)row * DIM_ + c] = (_Float16)((vals[i] - mu) * r * g[c] + bb[c]);
  }
}

// ---------------------------------------------------------------------------
// kv GEMM + mask + scatter. k -> [bh][Nk][64]; v -> transposed [bh][64][Nk].
// ---------------------------------------------------------------------------
__global__ __launch_bounds__(256) void kv_gemm(
    const _Float16* __restrict__ X, const _Float16* __restrict__ Wkv,
    const float* __restrict__ mask, _Float16* __restrict__ Kb,
    _Float16* __restrict__ Vt) {
  __shared__ __align__(16) _Float16 Wl[64 * LDW_];
  int lane = threadIdx.x & 31, wv = threadIdx.x >> 5;
  int mBase = blockIdx.x * 256 + wv * 32;
  int nBase = blockIdx.y * 64;

  STAGE_W(Wkv, DIM_)

  int koff = (lane >> 4) << 3;
  const _Float16* A0p = X + (size_t)(mBase + (lane & 15)) * DIM_ + koff;
  const _Float16* A1p = A0p + (size_t)16 * DIM_;
  GEMM_CORE(A0p, A1p, DIM_)

  int col0 = nBase + (lane & 15);
  int rhi  = (lane >> 4) << 3;
  int bB   = mBase >> 10;
#pragma unroll
  for (int s = 0; s < 2; ++s)
#pragma unroll
    for (int i = 0; i < 8; ++i) {
      int row = mBase + s * 16 + rhi + i;
      int ok  = row & 1023;
      float mv = mask[row];
#pragma unroll
      for (int t = 0; t < 4; ++t) {
        int n = col0 + t * 16;
        float val = acc[s][t][i] * mv;
        if (n < DIM_) {
          int bh = bB * HEADS_ + (n >> 6), d = n & 63;
          Kb[((size_t)bh * NK_ + ok) * HD_ + d] = (_Float16)val;
        } else {
          int n2 = n - DIM_;
          int bh = bB * HEADS_ + (n2 >> 6), d = n2 & 63;
          Vt[((size_t)bh * HD_ + d) * NK_ + ok] = (_Float16)val;
        }
      }
    }
}

// ---------------------------------------------------------------------------
// Flash attention with async double-buffered k/v chunks (32 keys).
// ---------------------------------------------------------------------------
__global__ __launch_bounds__(256) void attn_kernel(
    const _Float16* __restrict__ Q, const _Float16* __restrict__ Kb,
    const _Float16* __restrict__ Vt, _Float16* __restrict__ O) {
  __shared__ __align__(16) _Float16 kL[2][32 * 64];
  __shared__ __align__(16) _Float16 vL[2][64 * 32];
  __shared__ __align__(16) _Float16 pL[8 * 16 * 32];

  int lane = threadIdx.x & 31, wv = threadIdx.x >> 5;
  int bh = blockIdx.x, b = bh >> 3, h = bh & 7;
  int qBase = blockIdx.y * 128 + wv * 16;
  int koff  = (lane >> 4) << 3;

  const _Float16* qp =
      Q + ((size_t)(b * NQ_ + qBase + (lane & 15))) * DIM_ + h * HD_ + koff;
  v16h qa0 = frag16(qp);
  v16h qa1 = frag16(qp + 32);

  const _Float16* kbh = Kb + (size_t)bh * NK_ * HD_;
  const _Float16* vbh = Vt + (size_t)bh * HD_ * NK_;

  float m[8], l[8];
  v8f acc[4] = {{}, {}, {}, {}};
#pragma unroll
  for (int i = 0; i < 8; ++i) { m[i] = -3.0e38f; l[i] = 0.f; }

  const float scale = 0.125f;
  _Float16* myP = pL + wv * (16 * 32);

  int tkey = threadIdx.x >> 3, tkd = (threadIdx.x & 7) << 3;   // 32 x 64
  int tvd  = threadIdx.x >> 2, tvc = (threadIdx.x & 3) << 3;   // 64 x 32

  async_b128(kL[0] + tkey * 64 + tkd, kbh + (size_t)tkey * HD_ + tkd);
  async_b128(vL[0] + tvd * 32 + tvc, vbh + (size_t)tvd * NK_ + tvc);

  for (int it = 0; it < NK_ / 32; ++it) {
    int cur = it & 1;
    if (it + 1 < NK_ / 32) {
      int s1 = (it + 1) * 32;
      async_b128(kL[1 - cur] + tkey * 64 + tkd,
                 kbh + (size_t)(s1 + tkey) * HD_ + tkd);
      async_b128(vL[1 - cur] + tvd * 32 + tvc,
                 vbh + (size_t)tvd * NK_ + s1 + tvc);
      WAIT_ASYNC(2);
    } else {
      WAIT_ASYNC(0);
    }
    __syncthreads();

    const _Float16* kc_ = kL[cur];
    const _Float16* vc_ = vL[cur];

    // batch all 4 k fragments, then 4 WMMAs
    v16h kf[4];
#pragma unroll
    for (int j = 0; j < 2; ++j) {
      const _Float16* kp = kc_ + (j * 16 + (lane & 15)) * 64 + koff;
      kf[2 * j]     = frag16(kp);
      kf[2 * j + 1] = frag16(kp + 32);
    }
    v8f s[2];
#pragma unroll
    for (int j = 0; j < 2; ++j) {
      v8f sa = {};
      sa = wmma_f16(qa0, kf[2 * j], sa);
      sa = wmma_f16(qa1, kf[2 * j + 1], sa);
      s[j] = sa;
    }

    // online softmax (16-lane butterflies stay within lane halves)
#pragma unroll
    for (int i = 0; i < 8; ++i) {
      float v0 = s[0][i] * scale, v1 = s[1][i] * scale;
      float mx = fmaxf(v0, v1);
#pragma unroll
      for (int off = 1; off < 16; off <<= 1)
        mx = fmaxf(mx, __shfl_xor(mx, off, 32));
      float mn = fmaxf(m[i], mx);
      float a0 = __expf(v0 - mn), a1 = __expf(v1 - mn);
      float rs = a0 + a1;
#pragma unroll
      for (int off = 1; off < 16; off <<= 1) rs += __shfl_xor(rs, off, 32);
      float alpha = __expf(m[i] - mn);
      l[i] = l[i] * alpha + rs;
      m[i] = mn;
#pragma unroll
      for (int t = 0; t < 4; ++t) acc[t][i] *= alpha;
      int r = i + ((lane >> 4) << 3);
      myP[r * 32 + (lane & 15)]      = (_Float16)a0;
      myP[r * 32 + 16 + (lane & 15)] = (_Float16)a1;
    }
    __syncthreads();

    v16h pa = frag16(myP + (lane & 15) * 32 + koff);
    v16h vf[4];
#pragma unroll
    for (int t = 0; t < 4; ++t)
      vf[t] = frag16(vc_ + (t * 16 + (lane & 15)) * 32 + koff);
#pragma unroll
    for (int t = 0; t < 4; ++t)
      acc[t] = wmma_f16(pa, vf[t], acc[t]);
    __syncthreads();
  }

  int rhi = (lane >> 4) << 3;
#pragma unroll
  for (int t = 0; t < 4; ++t)
#pragma unroll
    for (int i = 0; i < 8; ++i) {
      int row = b * NQ_ + qBase + rhi + i;
      int col = h * HD_ + t * 16 + (lane & 15);
      O[(size_t)row * DIM_ + col] = (_Float16)(acc[t][i] / l[i]);
    }
}

// ---------------------------------------------------------------------------
// Output projection, f32 out + bias.
// ---------------------------------------------------------------------------
__global__ __launch_bounds__(256) void proj_gemm(
    const _Float16* __restrict__ A, const _Float16* __restrict__ W,
    const float* __restrict__ bias, float* __restrict__ O) {
  __shared__ __align__(16) _Float16 Wl[64 * LDW_];
  int lane = threadIdx.x & 31, wv = threadIdx.x >> 5;
  int mBase = blockIdx.x * 256 + wv * 32;
  int nBase = blockIdx.y * 64;

  STAGE_W(W, DIM_)

  int koff = (lane >> 4) << 3;
  const _Float16* A0p = A + (size_t)(mBase + (lane & 15)) * DIM_ + koff;
  const _Float16* A1p = A0p + (size_t)16 * DIM_;
  GEMM_CORE(A0p, A1p, DIM_)

  int col0 = nBase + (lane & 15);
  int rhi  = (lane >> 4) << 3;
#pragma unroll
  for (int t = 0; t < 4; ++t) {
    float bv = bias[col0 + t * 16];
#pragma unroll
    for (int s = 0; s < 2; ++s)
#pragma unroll
      for (int i = 0; i < 8; ++i)
        O[(size_t)(mBase + s * 16 + rhi + i) * DIM_ + col0 + t * 16] =
            acc[s][t][i] + bv;
  }
}

// ---------------------------------------------------------------------------
// Host launcher
// ---------------------------------------------------------------------------
extern "C" void kernel_launch(void* const* d_in, const int* in_sizes, int n_in,
                              void* d_out, int out_size, void* d_ws,
                              size_t ws_size, hipStream_t stream) {
  (void)in_sizes; (void)n_in; (void)out_size; (void)ws_size;
  const float* tokens = (const float*)d_in[0];
  const float* feats  = (const float*)d_in[1];
  const float* Wq     = (const float*)d_in[2];
  const float* Wkv    = (const float*)d_in[3];
  const float* sr_w   = (const float*)d_in[4];
  const float* sr_b   = (const float*)d_in[5];
  const float* ln_g   = (const float*)d_in[6];
  const float* ln_b   = (const float*)d_in[7];
  const float* Wp     = (const float*)d_in[8];
  const float* bp     = (const float*)d_in[9];
  const float* mask   = (const float*)d_in[10];
  float* out = (float*)d_out;

  char* ws = (char*)d_ws;
  size_t off = 0;
  auto carve = [&](size_t bytes) {
    char* p = ws + off;
    off += (bytes + 255) & ~(size_t)255;
    return p;
  };
  const size_t NTOK = (size_t)B_ * NQ_ * DIM_;
  const size_t NKV  = (size_t)B_ * HEADS_ * NK_ * HD_;
  _Float16* tokens16 = (_Float16*)carve(NTOK * 2);
  _Float16* feats16  = (_Float16*)carve(NTOK * 2);
  _Float16* Wq16     = (_Float16*)carve((size_t)DIM_ * DIM_ * 2);
  _Float16* Wkv16    = (_Float16*)carve((size_t)2 * DIM_ * DIM_ * 2);
  _Float16* Wp16     = (_Float16*)carve((size_t)DIM_ * DIM_ * 2);
  _Float16* Wsr16    = (_Float16*)carve((size_t)4 * DIM_ * DIM_ * 2);
  _Float16* q16      = (_Float16*)carve(NTOK * 2);
  float*    xconv    = (float*)   carve((size_t)B_ * NK_ * DIM_ * 4);
  _Float16* xln16    = (_Float16*)carve((size_t)B_ * NK_ * DIM_ * 2);
  _Float16* k16      = (_Float16*)carve(NKV * 2);
  _Float16* vT16     = (_Float16*)carve(NKV * 2);
  _Float16* attn16   = (_Float16*)carve(NTOK * 2);

  cvt_f32_to_f16<<<(NTOK + 255) / 256, 256, 0, stream>>>(tokens, tokens16, (int)NTOK);
  cvt_f32_to_f16<<<(NTOK + 255) / 256, 256, 0, stream>>>(feats, feats16, (int)NTOK);
  cvt_f32_to_f16<<<(DIM_ * DIM_ + 255) / 256, 256, 0, stream>>>(Wq, Wq16, DIM_ * DIM_);
  cvt_f32_to_f16<<<(2 * DIM_ * DIM_ + 255) / 256, 256, 0, stream>>>(Wkv, Wkv16, 2 * DIM_ * DIM_);
  cvt_f32_to_f16<<<(DIM_ * DIM_ + 255) / 256, 256, 0, stream>>>(Wp, Wp16, DIM_ * DIM_);
  repack_srw<<<(4 * DIM_ * DIM_ + 255) / 256, 256, 0, stream>>>(sr_w, Wsr16);

  gemm_f16out<<<dim3(B_ * NQ_ / 256, DIM_ / 64), 256, 0, stream>>>(
      tokens16, Wq16, q16, DIM_, DIM_);

  conv_gemm<<<dim3(B_ * NK_ / 256, DIM_ / 64), 256, 0, stream>>>(
      feats16, Wsr16, xconv);

  ln_kernel<<<B_ * NK_ / 8, 256, 0, stream>>>(xconv, sr_b, ln_g, ln_b, xln16);

  kv_gemm<<<dim3(B_ * NK_ / 256, 2 * DIM_ / 64), 256, 0, stream>>>(
      xln16, Wkv16, mask, k16, vT16);

  attn_kernel<<<dim3(B_ * HEADS_, NQ_ / 128), 256, 0, stream>>>(
      q16, k16, vT16, attn16);

  proj_gemm<<<dim3(B_ * NQ_ / 256, DIM_ / 64), 256, 0, stream>>>(
      attn16, Wp16, bp, out);
}
